// ASCA_0229_33998961115725
// MI455X (gfx1250) — matrix operand order repaired and checked by reading, not compiled, not run on hardware
//
#include <hip/hip_runtime.h>
#include <hip/hip_bf16.h>

typedef __attribute__((ext_vector_type(16))) _Float16 v16h;
typedef __attribute__((ext_vector_type(8)))  float    v8f;

#define NPIX   262144      // B*H*W
#define HW     65536       // H*W
#define NELEM  16777216    // B*C*H*W
#define DROW   2048        // W*c = H*c (attention row length)
#define EPS    1e-5f

__device__ __forceinline__ v8f wmma_f16(v16h a, v16h b, v8f c) {
  return __builtin_amdgcn_wmma_f32_16x16x32_f16(false, a, false, b, (short)0, c, false, false);
}

// ---------------- Generic 1x1 conv (64->64 channel GEMM) via WMMA ----------------
// in: f32 NCHW (B,64,H,W)  wgt: (O=64, K=64) row-major  out: f32 NCHW
__global__ __launch_bounds__(256) void k_conv1x1(const float* __restrict__ in,
                                                 const float* __restrict__ wgt,
                                                 const float* __restrict__ bias,
                                                 float* __restrict__ out) {
  int wave = blockIdx.x * (blockDim.x >> 5) + (threadIdx.x >> 5);
  int lane = threadIdx.x & 31;
  int p0   = wave * 16;                  // 16-pixel tile
  int m    = lane & 15;
  int hi8  = (lane & 16) ? 8 : 0;
  int hi16 = (lane & 16) ? 16 : 0;
  int pix  = p0 + m;
  int b    = pix >> 16;
  int hw   = pix & (HW - 1);
  const float* inp = in + (size_t)b * 64 * HW + hw;

  v16h a0, a1;                            // A tiles for k0=0 and k0=32
#pragma unroll
  for (int e = 0; e < 16; ++e) {
    int kk = e + hi16;
    a0[e] = (_Float16)inp[(size_t)kk * HW];
    a1[e] = (_Float16)inp[(size_t)(kk + 32) * HW];
  }
#pragma unroll
  for (int nt = 0; nt < 4; ++nt) {
    int n = nt * 16 + m;
    v16h b0, b1;
#pragma unroll
    for (int e = 0; e < 16; ++e) {
      b0[e] = (_Float16)wgt[n * 64 + hi16 + e];
      b1[e] = (_Float16)wgt[n * 64 + 32 + hi16 + e];
    }
    v8f acc = {};
    acc = wmma_f16(a0, b0, acc);
    acc = wmma_f16(a1, b1, acc);
    float bv = bias[n];
    float* op = out + (size_t)(b * 64 + n) * HW + (hw - m) + hi8;
#pragma unroll
    for (int r = 0; r < 8; ++r) op[r] = acc[r] + bv;
  }
}

// ---------------- BiasFree LayerNorm over channels ----------------
__global__ __launch_bounds__(256) void k_ln(const float* __restrict__ x,
                                            const float* __restrict__ lnw,
                                            float* __restrict__ y) {
  int p = blockIdx.x * blockDim.x + threadIdx.x;
  int b = p >> 16, hw = p & (HW - 1);
  const float* xp = x + (size_t)b * 64 * HW + hw;
  float s = 0.f, ss = 0.f;
#pragma unroll
  for (int c = 0; c < 64; ++c) { float v = xp[(size_t)c * HW]; s += v; ss += v * v; }
  float mean = s * (1.f / 64.f);
  float var  = ss * (1.f / 64.f) - mean * mean;
  float r    = rsqrtf(var + EPS);
  float* yp  = y + (size_t)b * 64 * HW + hw;
#pragma unroll
  for (int c = 0; c < 64; ++c) yp[(size_t)c * HW] = xp[(size_t)c * HW] * r * lnw[c];
}

// ---------------- Summed depthwise strip conv (3+7+11 taps) ----------------
__global__ __launch_bounds__(256) void k_strip(const float* __restrict__ x,
                                               const float* __restrict__ w3,  const float* __restrict__ b3,
                                               const float* __restrict__ w7,  const float* __restrict__ b7,
                                               const float* __restrict__ w11, const float* __restrict__ b11,
                                               float* __restrict__ y, int vertical) {
  int id = blockIdx.x * blockDim.x + threadIdx.x;
  int hw = id & (HW - 1); int bc = id >> 16; int c = bc & 63;
  int h = hw >> 8, w = hw & 255;
  float taps[11];
#pragma unroll
  for (int t = 0; t < 11; ++t) {
    float v = w11[c * 11 + t];
    int t7 = t - 2; if (t7 >= 0 && t7 < 7) v += w7[c * 7 + t7];
    int t3 = t - 4; if (t3 >= 0 && t3 < 3) v += w3[c * 3 + t3];
    taps[t] = v;
  }
  float acc = b3[c] + b7[c] + b11[c];
  const float* xp = x + ((size_t)bc << 16);
  if (vertical) {
#pragma unroll
    for (int t = 0; t < 11; ++t) { int hh = h + t - 5; if (hh >= 0 && hh < 256) acc += taps[t] * xp[hh * 256 + w]; }
  } else {
#pragma unroll
    for (int t = 0; t < 11; ++t) { int ww = w + t - 5; if (ww >= 0 && ww < 256) acc += taps[t] * xp[h * 256 + ww]; }
  }
  y[id] = acc;
}

// ---- Pack NCHW f32 -> f16 attention layouts ----
// vt==0: row layout  dst[bh*256 + seq][feature]   (rows of length 2048, +optional L2 norm)
// vt==1: transposed  dst[bh][feature][seq]        (feature-major; for V so WMMA B-operand
//                                                  loads become contiguous 32B runs per lane)
__global__ __launch_bounds__(256) void k_pack(const float* __restrict__ src,
                                              _Float16* __restrict__ dst,
                                              int wh, int norm, int vt) {
  __shared__ float red[256];
  int row  = blockIdx.x;                 // b*2048 + head*256 + (h|w)
  int b    = row >> 11;
  int head = (row >> 8) & 7;
  int i    = row & 255;                  // sequence coordinate
  int t    = threadIdx.x;                // other spatial coord
  float v[8]; float ss = 0.f;
#pragma unroll
  for (int ci = 0; ci < 8; ++ci) {
    size_t addr = (((size_t)(b * 64 + head * 8 + ci)) << 16) + (wh ? (t * 256 + i) : (i * 256 + t));
    v[ci] = src[addr]; ss += v[ci] * v[ci];
  }
  red[t] = ss; __syncthreads();
  for (int o = 128; o > 0; o >>= 1) { if (t < o) red[t] += red[t + o]; __syncthreads(); }
  float scale = 1.f;
  if (norm) scale = 1.f / fmaxf(sqrtf(red[0]), 1e-12f);
  if (vt) {
    int bh = row >> 8;
    _Float16* dp = dst + (size_t)bh * 256 * DROW + (size_t)(t * 8) * 256 + i;
#pragma unroll
    for (int ci = 0; ci < 8; ++ci) dp[(size_t)ci * 256] = (_Float16)(v[ci] * scale);
  } else {
    _Float16* dp = dst + (size_t)row * DROW + t * 8;
#pragma unroll
    for (int ci = 0; ci < 8; ++ci) dp[ci] = (_Float16)(v[ci] * scale);
  }
}

// ---------------- Attention: softmax(q k^T) v + q, merged back into NCHW accumulator ----------------
// q,k: row layout [bh][seq][feature];  vtr: transposed layout [bh][feature][seq]
__global__ __launch_bounds__(64) void k_attn(const _Float16* __restrict__ q,
                                             const _Float16* __restrict__ k,
                                             const _Float16* __restrict__ vtr,
                                             float* __restrict__ acc_out, int merge_wh) {
  __shared__ _Float16 S[2][16 * 256];    // per-wave score tile, f16 (A-operand ready)
  int wv   = threadIdx.x >> 5;
  int lane = threadIdx.x & 31;
  int tile = blockIdx.x * 2 + wv;        // 0..511
  int bh   = tile >> 4;                  // (b,head) 0..31
  int m0   = (tile & 15) * 16;           // query-row tile base
  int m    = lane & 15;
  int hi8  = (lane & 16) ? 8 : 0;
  int hi16 = (lane & 16) ? 16 : 0;
  const _Float16* qb = q   + (size_t)bh * 256 * DROW;
  const _Float16* kb = k   + (size_t)bh * 256 * DROW;
  const _Float16* vb = vtr + (size_t)bh * 256 * DROW;

  // ---- Stage 1: S = q @ k^T  (16 x 256 in 16 f32 accumulator octets) ----
  v8f s[16];
#pragma unroll
  for (int j = 0; j < 16; ++j) s[j] = (v8f){};
  for (int k0 = 0; k0 < DROW; k0 += 32) {
    // warm V^T panel into cache for stage 2 while the WMMAs run
    __builtin_prefetch((const char*)vb + ((size_t)(k0 >> 5) * 32 + lane) * 512, 0, 1);
    v16h a;
    const _Float16* qr = qb + (size_t)(m0 + m) * DROW + k0 + hi8;
#pragma unroll
    for (int e = 0; e < 16; ++e) a[e] = qr[e + hi8];
#pragma unroll
    for (int j = 0; j < 16; ++j) {
      v16h bt;
      const _Float16* kr = kb + (size_t)(j * 16 + m) * DROW + k0 + hi16;
#pragma unroll
      for (int e = 0; e < 16; ++e) bt[e] = kr[e];
      s[j] = wmma_f16(a, bt, s[j]);
    }
  }

  // ---- Softmax over rows (row M = r + hi8; elements across 16 lanes x 16 tiles) ----
#pragma unroll
  for (int r = 0; r < 8; ++r) {
    float mx = s[0][r];
#pragma unroll
    for (int j = 1; j < 16; ++j) mx = fmaxf(mx, s[j][r]);
#pragma unroll
    for (int o = 1; o < 16; o <<= 1) mx = fmaxf(mx, __shfl_xor(mx, o, 32));
    float sum = 0.f;
#pragma unroll
    for (int j = 0; j < 16; ++j) { float e = __expf(s[j][r] - mx); s[j][r] = e; sum += e; }
#pragma unroll
    for (int o = 1; o < 16; o <<= 1) sum += __shfl_xor(sum, o, 32);
    float inv = 1.f / sum;
#pragma unroll
    for (int j = 0; j < 16; ++j) s[j][r] *= inv;
  }

  // ---- Round-trip S through LDS (convert to f16 ONCE; re-swizzle D-layout -> A-layout) ----
  _Float16* Sw = S[wv];
#pragma unroll
  for (int j = 0; j < 16; ++j)
#pragma unroll
    for (int r = 0; r < 8; ++r)
      Sw[(r + hi8) * 256 + j * 16 + m] = (_Float16)s[j][r];
  __syncthreads();

  // ---- Stage 2: O = S @ V + q, merge back into NCHW ----
  int head = bh & 7, bb = bh >> 3;
  for (int jn = 0; jn < 128; ++jn) {
    v8f o = {};
#pragma unroll
    for (int k0 = 0; k0 < 256; k0 += 32) {
      v16h a;
      const _Float16* sr = Sw + m * 256 + k0 + hi8;
#pragma unroll
      for (int e = 0; e < 16; ++e) a[e] = sr[e + hi8];
      v16h bt;
      const _Float16* vr = vb + (size_t)(jn * 16 + m) * 256 + k0 + hi16;   // contiguous 32B run
#pragma unroll
      for (int e = 0; e < 16; ++e) bt[e] = vr[e];
      o = wmma_f16(a, bt, o);
    }
#pragma unroll
    for (int r = 0; r < 8; ++r) {
      int row = m0 + r + hi8;            // h (hw-merge) or w (wh-merge)
      int col = jn * 16 + m;             // w*8+ci or h*8+ci
      float val = o[r] + (float)qb[(size_t)row * DROW + col];
      int sp = col >> 3, ci = col & 7;
      size_t addr = ((size_t)(bb * 64 + head * 8 + ci)) << 16;
      addr += merge_wh ? (size_t)(sp * 256 + row) : (size_t)(row * 256 + sp);
      acc_out[addr] += val;              // unique writer per element; kernels serialized
    }
  }
}

// ---------------- BatchNorm (training stats) + ReLU ----------------
__global__ __launch_bounds__(256) void k_bnstat1(const float* __restrict__ x,
                                                 float* __restrict__ psum, float* __restrict__ pss) {
  __shared__ float rs[256], rq[256];
  int ch = blockIdx.x >> 6, chunk = blockIdx.x & 63;
  int t = threadIdx.x;
  float s = 0.f, q = 0.f;
  int i0 = chunk * 4096;
  for (int j = 0; j < 16; ++j) {
    int i = i0 + j * 256 + t;            // 0..262143 within channel
    int b = i >> 16, hw = i & (HW - 1);
    float v = x[(((size_t)(b * 64 + ch)) << 16) + hw];
    s += v; q += v * v;
  }
  rs[t] = s; rq[t] = q; __syncthreads();
  for (int o = 128; o > 0; o >>= 1) { if (t < o) { rs[t] += rs[t + o]; rq[t] += rq[t + o]; } __syncthreads(); }
  if (t == 0) { psum[blockIdx.x] = rs[0]; pss[blockIdx.x] = rq[0]; }
}

__global__ __launch_bounds__(64) void k_bnstat2(const float* __restrict__ psum, const float* __restrict__ pss,
                                                float* __restrict__ mu, float* __restrict__ var) {
  __shared__ float rs[64], rq[64];
  int ch = blockIdx.x, t = threadIdx.x;
  rs[t] = psum[ch * 64 + t]; rq[t] = pss[ch * 64 + t]; __syncthreads();
  for (int o = 32; o > 0; o >>= 1) { if (t < o) { rs[t] += rs[t + o]; rq[t] += rq[t + o]; } __syncthreads(); }
  if (t == 0) { float mval = rs[0] / 262144.f; mu[ch] = mval; var[ch] = rq[0] / 262144.f - mval * mval; }
}

__global__ __launch_bounds__(256) void k_bnapply(const float* __restrict__ x,
                                                 const float* __restrict__ mu, const float* __restrict__ var,
                                                 const float* __restrict__ g,  const float* __restrict__ bb,
                                                 float* __restrict__ out) {
  size_t id = (size_t)blockIdx.x * blockDim.x + threadIdx.x;
  int c = (int)((id >> 16) & 63);
  float vv = (x[id] - mu[c]) * rsqrtf(var[c] + EPS) * g[c] + bb[c];
  out[id] = fmaxf(vv, 0.f);
}

// ---------------- Host orchestration ----------------
extern "C" void kernel_launch(void* const* d_in, const int* in_sizes, int n_in,
                              void* d_out, int out_size, void* d_ws, size_t ws_size,
                              hipStream_t stream) {
  const float* x_in  = (const float*)d_in[0];
  const float* w_in  = (const float*)d_in[1];
  const float* b_in  = (const float*)d_in[2];
  const float* ln_w  = (const float*)d_in[3];
  const float* dw01w = (const float*)d_in[4];  const float* dw01b = (const float*)d_in[5];
  const float* dw02w = (const float*)d_in[6];  const float* dw02b = (const float*)d_in[7];
  const float* dw11w = (const float*)d_in[8];  const float* dw11b = (const float*)d_in[9];
  const float* dw12w = (const float*)d_in[10]; const float* dw12b = (const float*)d_in[11];
  const float* dw21w = (const float*)d_in[12]; const float* dw21b = (const float*)d_in[13];
  const float* dw22w = (const float*)d_in[14]; const float* dw22b = (const float*)d_in[15];
  const float* wq1 = (const float*)d_in[16]; const float* bq1 = (const float*)d_in[17];
  const float* wq2 = (const float*)d_in[18]; const float* bq2 = (const float*)d_in[19];
  const float* wk1 = (const float*)d_in[20]; const float* bk1 = (const float*)d_in[21];
  const float* wk2 = (const float*)d_in[22]; const float* bk2 = (const float*)d_in[23];
  const float* wv1 = (const float*)d_in[24]; const float* bv1 = (const float*)d_in[25];
  const float* wv2 = (const float*)d_in[26]; const float* bv2 = (const float*)d_in[27];
  const float* w_out = (const float*)d_in[28]; const float* b_out = (const float*)d_in[29];
  const float* bn_g  = (const float*)d_in[30]; const float* bn_b  = (const float*)d_in[31];
  float* out = (float*)d_out;

  const size_t NB = NELEM;
  float* W0 = (float*)d_ws;        // conv_in output, then attention accumulator (+x residual)
  float* W1 = W0 + NB;             // LN output, then projection scratch, then conv_out output
  float* W2 = W1 + NB;             // out1
  float* W3 = W2 + NB;             // out2
  _Float16* F16  = (_Float16*)(W3 + NB);
  _Float16* q1hw = F16 + 0 * NB;   // l2n(split_hw(q1))
  _Float16* q1wh = F16 + 1 * NB;   // l2n(split_wh(q1))
  _Float16* q2hw = F16 + 2 * NB;   // l2n(split_hw(q2))
  _Float16* q2wh = F16 + 3 * NB;   // l2n(split_wh(q2))
  _Float16* k1h  = F16 + 4 * NB;   // l2n(split_hw(k1))
  _Float16* k2w  = F16 + 5 * NB;   // l2n(split_wh(k2))
  _Float16* v1t  = F16 + 6 * NB;   // split_hw(v1), feature-major (transposed)
  _Float16* v2t  = F16 + 7 * NB;   // split_wh(v2), feature-major (transposed)
  float* psum = (float*)(F16 + 8 * NB);
  float* pss  = psum + 4096;
  float* mu   = pss + 4096;
  float* var  = mu + 64;

  dim3 blk256(256), blk64(64);

  // 1. x = conv1x1(x_in, w_in) + b_in
  k_conv1x1<<<2048, blk256, 0, stream>>>(x_in, w_in, b_in, W0);
  // 2. x1 = biasfree_ln(x)
  k_ln<<<1024, blk256, 0, stream>>>(W0, ln_w, W1);
  // 3. out1 (horizontal strips), out2 (vertical strips)
  k_strip<<<65536, blk256, 0, stream>>>(W1, dw01w, dw01b, dw11w, dw11b, dw21w, dw21b, W2, 0);
  k_strip<<<65536, blk256, 0, stream>>>(W1, dw02w, dw02b, dw12w, dw12b, dw22w, dw22b, W3, 1);
  // 4. projections -> packed f16 layouts
  k_conv1x1<<<2048, blk256, 0, stream>>>(W2, wq1, bq1, W1);
  k_pack<<<8192, blk256, 0, stream>>>(W1, q1hw, 0, 1, 0);
  k_pack<<<8192, blk256, 0, stream>>>(W1, q1wh, 1, 1, 0);
  k_conv1x1<<<2048, blk256, 0, stream>>>(W2, wk1, bk1, W1);
  k_pack<<<8192, blk256, 0, stream>>>(W1, k1h, 0, 1, 0);
  k_conv1x1<<<2048, blk256, 0, stream>>>(W2, wv1, bv1, W1);
  k_pack<<<8192, blk256, 0, stream>>>(W1, v1t, 0, 0, 1);
  k_conv1x1<<<2048, blk256, 0, stream>>>(W3, wq2, bq2, W1);
  k_pack<<<8192, blk256, 0, stream>>>(W1, q2hw, 0, 1, 0);
  k_pack<<<8192, blk256, 0, stream>>>(W1, q2wh, 1, 1, 0);
  k_conv1x1<<<2048, blk256, 0, stream>>>(W3, wk2, bk2, W1);
  k_pack<<<8192, blk256, 0, stream>>>(W1, k2w, 1, 1, 0);
  k_conv1x1<<<2048, blk256, 0, stream>>>(W3, wv2, bv2, W1);
  k_pack<<<8192, blk256, 0, stream>>>(W1, v2t, 1, 0, 1);
  // 5. four attentions accumulate into W0 (which already holds x -> residual)
  k_attn<<<256, blk64, 0, stream>>>(q1hw, k1h, v1t, W0, 0);
  k_attn<<<256, blk64, 0, stream>>>(q2wh, k2w, v2t, W0, 1);
  k_attn<<<256, blk64, 0, stream>>>(q2hw, k1h, v1t, W0, 0);
  k_attn<<<256, blk64, 0, stream>>>(q1wh, k2w, v2t, W0, 1);
  // 6. out = conv1x1(W0, w_out) + b_out
  k_conv1x1<<<2048, blk256, 0, stream>>>(W0, w_out, b_out, W1);
  // 7. BatchNorm (batch stats) + ReLU
  k_bnstat1<<<4096, blk256, 0, stream>>>(W1, psum, pss);
  k_bnstat2<<<64, blk64, 0, stream>>>(psum, pss, mu, var);
  k_bnapply<<<65536, blk256, 0, stream>>>(W1, mu, var, bn_g, bn_b, out);
}
